// MultiHeadAttentionClassifier_87505663688834
// MI455X (gfx1250) — matrix operand order repaired and checked by reading, not compiled, not run on hardware
//
#include <hip/hip_runtime.h>
#include <hip/hip_bf16.h>

// ---------------------------------------------------------------------------
// MI455X (gfx1250) implementation: CBAM + 8-head self-attention + classifier
// Heavy GEMMs: v_wmma_f32_16x16x32_bf16 (f32 accumulate).
// Flash attention stages K/V tiles into LDS with global_load_async_to_lds_b128
// (ASYNCcnt / s_wait_asynccnt), double-buffered, 8 waves per workgroup.
// ---------------------------------------------------------------------------

typedef __bf16 bf16;
typedef __attribute__((ext_vector_type(16))) __bf16 v16bf;
typedef __attribute__((ext_vector_type(8)))  float   v8f;

#define BSZ   16
#define CCH   512
#define HW    28
#define NTOK  784      // 28*28
#define NHEAD 8
#define HDIM  64
#define NPAD  800      // padded key stride for V^T (multiple of 32, >= 784+16)
#define KTILES 25      // ceil(784/32); last tile has 16 valid keys

union Frag16 { v16bf v; uint4 q[2]; };

// A-fragment: 16x32 bf16, row-major source, stride lda (elements).
// lane L holds row L%16; K chunks at k0=(L<16?0:8) and k0+16. (ISA 7.12.2)
__device__ __forceinline__ v16bf load_a16(const bf16* base, int lda) {
  int lane = threadIdx.x & 31;
  int row  = lane & 15;
  int k0   = (lane & 16) >> 1;               // 0 or 8
  const bf16* p = base + row * lda + k0;
  Frag16 f;
  f.q[0] = *reinterpret_cast<const uint4*>(p);
  f.q[1] = *reinterpret_cast<const uint4*>(p + 16);
  return f.v;
}

// B-fragment: 32x16 bf16 where column n is a contiguous row of `bt`
// (i.e. bt is B^T row-major, stride ldb). lane n: col n%16, K chunks at k0=(n<16?0:8) and k0+16.
__device__ __forceinline__ v16bf load_b16(const bf16* bt, int ldb) {
  int lane = threadIdx.x & 31;
  int col  = lane & 15;
  int k0   = (lane & 16) >> 1;               // 0 or 8
  const bf16* p = bt + col * ldb + k0;
  Frag16 f;
  f.q[0] = *reinterpret_cast<const uint4*>(p);
  f.q[1] = *reinterpret_cast<const uint4*>(p + 16);
  return f.v;
}

__device__ __forceinline__ v8f wmma_bf16(v16bf a, v16bf b, v8f c) {
  return __builtin_amdgcn_wmma_f32_16x16x32_bf16(false, a, false, b,
                                                 (short)0, c, false, false);
}

// Async global -> LDS, 16 bytes per lane (ASYNCcnt path).
__device__ __forceinline__ void async_copy16(unsigned lds_addr, const void* g) {
  unsigned long long ga = (unsigned long long)g;
  asm volatile("global_load_async_to_lds_b128 %0, %1, off"
               :: "v"(lds_addr), "v"(ga) : "memory");
}
__device__ __forceinline__ void wait_async_le2() {
  asm volatile("s_wait_asynccnt 0x2" ::: "memory");
}
__device__ __forceinline__ void wait_async_0() {
  asm volatile("s_wait_asynccnt 0x0" ::: "memory");
}
__device__ __forceinline__ unsigned lds_off(const void* p) {
  // LDS aperture keeps the wave-relative byte offset in addr[31:0]
  return (unsigned)(unsigned long long)p;
}

// ---------------------------------------------------------------------------
// K0: f32 -> bf16 conversion
// ---------------------------------------------------------------------------
__global__ void cvt_f32_bf16(const float* __restrict__ s, bf16* __restrict__ d, int n) {
  int i = blockIdx.x * 256 + threadIdx.x;
  if (i < n) d[i] = (bf16)s[i];
}

// ---------------------------------------------------------------------------
// K1: per-(b,c) avg & max pool over HW
// ---------------------------------------------------------------------------
__global__ void pool_avgmax(const float* __restrict__ x,
                            float* __restrict__ avg, float* __restrict__ mx) {
  int bc = blockIdx.x;
  const float* p = x + (size_t)bc * NTOK;
  __shared__ float ssum[256], smax[256];
  int tid = threadIdx.x;
  float s = 0.f, m = -1e30f;
  for (int i = tid; i < NTOK; i += 256) { float v = p[i]; s += v; m = fmaxf(m, v); }
  ssum[tid] = s; smax[tid] = m; __syncthreads();
  for (int st = 128; st > 0; st >>= 1) {
    if (tid < st) { ssum[tid] += ssum[tid + st]; smax[tid] = fmaxf(smax[tid], smax[tid + st]); }
    __syncthreads();
  }
  if (tid == 0) { avg[bc] = ssum[0] * (1.f / NTOK); mx[bc] = smax[0]; }
}

// ---------------------------------------------------------------------------
// K2: channel gate MLP: sigmoid(mlp(avg)+mlp(max)) per (b,c)
// ---------------------------------------------------------------------------
__global__ void channel_mlp(const float* __restrict__ avg, const float* __restrict__ mxp,
                            const float* __restrict__ w1, const float* __restrict__ b1,
                            const float* __restrict__ w2, const float* __restrict__ b2,
                            float* __restrict__ ch) {
  int b = blockIdx.x, tid = threadIdx.x;
  __shared__ float sa[CCH], sm[CCH], hid[32];
  for (int c = tid; c < CCH; c += 256) { sa[c] = avg[b * CCH + c]; sm[c] = mxp[b * CCH + c]; }
  __syncthreads();
  if (tid < 32) {
    float aa = b1[tid], mm = b1[tid];
    for (int c = 0; c < CCH; c++) { float w = w1[tid * CCH + c]; aa += w * sa[c]; mm += w * sm[c]; }
    hid[tid] = fmaxf(aa, 0.f) + fmaxf(mm, 0.f);
  }
  __syncthreads();
  for (int c = tid; c < CCH; c += 256) {
    float acc = 2.f * b2[c];
    for (int r = 0; r < 32; r++) acc += hid[r] * w2[c * 32 + r];
    ch[b * CCH + c] = 1.f / (1.f + __expf(-acc));
  }
}

// ---------------------------------------------------------------------------
// K3: apply channel scale, produce channel-max/mean maps
// ---------------------------------------------------------------------------
__global__ void scale_comp(const float* __restrict__ x, const float* __restrict__ ch,
                           float* __restrict__ xg,
                           float* __restrict__ cmax, float* __restrict__ cmean) {
  int b = blockIdx.y;
  int n = blockIdx.x * 256 + threadIdx.x;
  if (n >= NTOK) return;
  float mx = -1e30f, sm = 0.f;
  size_t base = (size_t)b * CCH * NTOK + n;
  for (int c = 0; c < CCH; c++) {
    float v = x[base + (size_t)c * NTOK] * ch[b * CCH + c];
    xg[base + (size_t)c * NTOK] = v;
    mx = fmaxf(mx, v); sm += v;
  }
  cmax[b * NTOK + n] = mx;
  cmean[b * NTOK + n] = sm * (1.f / CCH);
}

// ---------------------------------------------------------------------------
// K4: 7x7 spatial conv (2->1) + BN + sigmoid
// ---------------------------------------------------------------------------
__global__ void spatial_sig(const float* __restrict__ cmax, const float* __restrict__ cmean,
                            const float* __restrict__ w, const float* __restrict__ gam,
                            const float* __restrict__ bet, const float* __restrict__ mean,
                            const float* __restrict__ var, float* __restrict__ sig) {
  int b = blockIdx.y;
  int n = blockIdx.x * 256 + threadIdx.x;
  if (n >= NTOK) return;
  int y = n / HW, x0 = n % HW;
  float acc = 0.f;
  for (int i = 0; i < 7; i++) {
    int yy = y + i - 3; if (yy < 0 || yy >= HW) continue;
    for (int j = 0; j < 7; j++) {
      int xx = x0 + j - 3; if (xx < 0 || xx >= HW) continue;
      int idx = b * NTOK + yy * HW + xx;
      acc += w[i * 7 + j] * cmax[idx] + w[49 + i * 7 + j] * cmean[idx];
    }
  }
  float s = (acc - mean[0]) * rsqrtf(var[0] + 1e-5f) * gam[0] + bet[0];
  sig[b * NTOK + n] = 1.f / (1.f + __expf(-s));
}

// ---------------------------------------------------------------------------
// K5: apply spatial gate in place (f32 [B,C,N]) and transpose to bf16 [B,N,C]
// ---------------------------------------------------------------------------
__global__ void gate_transpose(float* __restrict__ xg, const float* __restrict__ sig,
                               bf16* __restrict__ xbf) {
  int b = blockIdx.z, c0 = blockIdx.y * 16, n0 = blockIdx.x * 16;
  int tx = threadIdx.x & 15, ty = threadIdx.x >> 4;
  __shared__ float t[16][17];
  size_t idx = ((size_t)b * CCH + c0 + ty) * NTOK + n0 + tx;
  float v = xg[idx] * sig[b * NTOK + n0 + tx];
  xg[idx] = v;
  t[ty][tx] = v;
  __syncthreads();
  xbf[((size_t)b * NTOK + n0 + ty) * CCH + c0 + tx] = (bf16)t[tx][ty];
}

// ---------------------------------------------------------------------------
// K6: fused QKV projection, WMMA bf16. Q,K -> [B,N,C]; V -> V^T [B,H,d,NPAD]
// ---------------------------------------------------------------------------
__global__ __launch_bounds__(256) void qkv_proj(
    const bf16* __restrict__ xbf,
    const bf16* __restrict__ wqb, const bf16* __restrict__ wkb, const bf16* __restrict__ wvb,
    const float* __restrict__ bq, const float* __restrict__ bk, const float* __restrict__ bv,
    bf16* __restrict__ Q, bf16* __restrict__ K, bf16* __restrict__ Vt) {
  int b = blockIdx.z;
  int n0 = blockIdx.x * 16;
  int ct = blockIdx.y * 8 + (threadIdx.x >> 5);
  int lane = threadIdx.x & 31, col = lane & 15, rbase = (lane & 16) >> 1;
  const bf16* arow = xbf + ((size_t)b * NTOK + n0) * CCH;
  v8f aq = {0,0,0,0,0,0,0,0}, ak = aq, av = aq;
  for (int ks = 0; ks < CCH / 32; ks++) {
    v16bf a = load_a16(arow + ks * 32, CCH);
    aq = wmma_bf16(a, load_b16(wqb + ct * 16 * CCH + ks * 32, CCH), aq);
    ak = wmma_bf16(a, load_b16(wkb + ct * 16 * CCH + ks * 32, CCH), ak);
    av = wmma_bf16(a, load_b16(wvb + ct * 16 * CCH + ks * 32, CCH), av);
  }
  int cout = ct * 16 + col;
  float bqv = bq[cout], bkv = bk[cout], bvv = bv[cout];
  int h = cout >> 6, d = cout & 63;
  for (int r = 0; r < 8; r++) {
    int n = n0 + rbase + r;
    Q[((size_t)b * NTOK + n) * CCH + cout] = (bf16)(aq[r] + bqv);
    K[((size_t)b * NTOK + n) * CCH + cout] = (bf16)(ak[r] + bkv);
    Vt[(((size_t)b * NHEAD + h) * HDIM + d) * NPAD + n] = (bf16)(av[r] + bvv);
  }
}

// ---------------------------------------------------------------------------
// K7: flash attention.
// Block = 8 waves; wave w handles query tile (blockIdx.x*8 + w) for (b,h).
// K/V tiles (32 keys) staged in LDS via async loads, double-buffered.
// ---------------------------------------------------------------------------
__global__ __launch_bounds__(256) void flash_attn(
    const bf16* __restrict__ Q, const bf16* __restrict__ K,
    const bf16* __restrict__ Vt, bf16* __restrict__ O) {
  __shared__ __align__(16) bf16 sK[2][32 * 64];   // [buf][key][d]
  __shared__ __align__(16) bf16 sV[2][64 * 32];   // [buf][d][key]
  __shared__ __align__(16) bf16 sP[8][16 * 32];   // per-wave P staging

  int h = blockIdx.y, b = blockIdx.z;
  int tid = threadIdx.x;
  int wave = tid >> 5;
  int lane = tid & 31, col = lane & 15, rbase = (lane & 16) >> 1;
  int qtile = blockIdx.x * 8 + wave;
  bool active = qtile < (NTOK / 16);
  int qt = active ? qtile : (NTOK / 16 - 1);    // clamp: duplicate work, no store
  int n0 = qt * 16;

  const bf16* Qb = Q + ((size_t)b * NTOK) * CCH + h * HDIM;
  const bf16* Kb = K + ((size_t)b * NTOK) * CCH + h * HDIM;
  const bf16* Vb = Vt + ((size_t)b * NHEAD + h) * HDIM * NPAD;

  v16bf aq0 = load_a16(Qb + (size_t)n0 * CCH, CCH);
  v16bf aq1 = load_a16(Qb + (size_t)n0 * CCH + 32, CCH);

  // cooperative async tile fetch: K tile 32x64 bf16 (4KB), V tile 64x32 (4KB)
  int krow = tid >> 3, kchk = tid & 7;     // 32 rows x 8 chunks of 16B
  int vrow = tid >> 2, vchk = tid & 3;     // 64 rows x 4 chunks of 16B
  auto issue = [&](int m0, int buf) {
    async_copy16(lds_off(&sK[buf][krow * 64]) + kchk * 16,
                 (const char*)(Kb + (size_t)(m0 + krow) * CCH) + kchk * 16);
    async_copy16(lds_off(&sV[buf][vrow * 32]) + vchk * 16,
                 (const char*)(Vb + (size_t)vrow * NPAD + m0) + vchk * 16);
  };

  const v8f vzero = {0,0,0,0,0,0,0,0};
  float mrow[8], lrow[8];
  v8f oacc[4];
#pragma unroll
  for (int r = 0; r < 8; r++) { mrow[r] = -1e30f; lrow[r] = 0.f; }
#pragma unroll
  for (int t = 0; t < 4; t++) oacc[t] = vzero;

  bf16* Pw = &sP[wave][0];

  issue(0, 0);
  for (int t = 0; t < KTILES; ++t) {
    int m0 = t * 32;
    int cur = t & 1;
    bool more = (t + 1) < KTILES;
    if (more) issue(m0 + 32, 1 - cur);
    if (more) wait_async_le2(); else wait_async_0();
    __syncthreads();                       // all waves' tile parts visible

    bool has2 = (m0 + 16) < NTOK;
    const bf16* kt = &sK[cur][0];
    const bf16* vt = &sV[cur][0];
    v8f s0 = vzero, s1 = vzero;
    s0 = wmma_bf16(aq0, load_b16(kt, 64), s0);
    s0 = wmma_bf16(aq1, load_b16(kt + 32, 64), s0);
    if (has2) {
      s1 = wmma_bf16(aq0, load_b16(kt + 16 * 64, 64), s1);
      s1 = wmma_bf16(aq1, load_b16(kt + 16 * 64 + 32, 64), s1);
    }
#pragma unroll
    for (int r = 0; r < 8; r++) {
      float a0 = s0[r] * 0.125f;                       // 1/sqrt(64)
      float a1 = has2 ? s1[r] * 0.125f : -1e30f;
      float tmax = fmaxf(a0, a1);
      for (int off = 1; off < 16; off <<= 1) tmax = fmaxf(tmax, __shfl_xor(tmax, off, 32));
      float mnew  = fmaxf(mrow[r], tmax);
      float alpha = __expf(mrow[r] - mnew);
      mrow[r] = mnew;
      float p0 = __expf(a0 - mnew);
      float p1 = has2 ? __expf(a1 - mnew) : 0.f;
      float rs = p0 + p1;
      for (int off = 1; off < 16; off <<= 1) rs += __shfl_xor(rs, off, 32);
      lrow[r] = lrow[r] * alpha + rs;
      oacc[0][r] *= alpha; oacc[1][r] *= alpha; oacc[2][r] *= alpha; oacc[3][r] *= alpha;
      Pw[(rbase + r) * 32 + col]      = (bf16)p0;
      Pw[(rbase + r) * 32 + 16 + col] = (bf16)p1;
    }
    v16bf ap = load_a16(Pw, 32);           // wave-private LDS, DS ops in order
#pragma unroll
    for (int d = 0; d < 4; d++)
      oacc[d] = wmma_bf16(ap, load_b16(vt + d * 16 * 32, 32), oacc[d]);
    __syncthreads();                       // done reading buf `cur` everywhere
  }

  if (active) {
#pragma unroll
    for (int d = 0; d < 4; d++)
      for (int r = 0; r < 8; r++) {
        float v = oacc[d][r] / lrow[r];
        O[((size_t)b * NTOK + n0 + rbase + r) * CCH + h * HDIM + d * 16 + col] = (bf16)v;
      }
  }
}

// ---------------------------------------------------------------------------
// K8: output projection (WMMA) + bias + residual + pooled sum (atomic)
// ---------------------------------------------------------------------------
__global__ __launch_bounds__(256) void out_proj(
    const bf16* __restrict__ A, const bf16* __restrict__ wob,
    const float* __restrict__ bo, const float* __restrict__ xg,
    float* __restrict__ pooled) {
  int b = blockIdx.z;
  int n0 = blockIdx.x * 16;
  int ct = blockIdx.y * 8 + (threadIdx.x >> 5);
  int lane = threadIdx.x & 31, col = lane & 15, rbase = (lane & 16) >> 1;
  const bf16* arow = A + ((size_t)b * NTOK + n0) * CCH;
  v8f acc = {0,0,0,0,0,0,0,0};
  for (int ks = 0; ks < CCH / 32; ks++)
    acc = wmma_bf16(load_a16(arow + ks * 32, CCH),
                    load_b16(wob + ct * 16 * CCH + ks * 32, CCH), acc);
  int cout = ct * 16 + col;
  float bb = bo[cout];
  float part = 0.f;
  for (int r = 0; r < 8; r++) {
    int n = n0 + rbase + r;
    part += acc[r] + bb + xg[((size_t)b * CCH + cout) * NTOK + n];
  }
  atomicAdd(&pooled[b * CCH + cout], part);
}

// ---------------------------------------------------------------------------
// K9: LayerNorm -> fc1+ReLU -> fc2 (one block per batch row)
// ---------------------------------------------------------------------------
__global__ void classifier(const float* __restrict__ pooled,
                           const float* __restrict__ lng, const float* __restrict__ lnb,
                           const float* __restrict__ w1, const float* __restrict__ b1,
                           const float* __restrict__ w2, const float* __restrict__ b2,
                           float* __restrict__ out) {
  int b = blockIdx.x, tid = threadIdx.x;
  __shared__ float pl[CCH], red[256], hb[CCH];
  float s = 0.f, s2 = 0.f;
  for (int c = tid; c < CCH; c += 256) {
    float v = pooled[b * CCH + c] * (1.f / NTOK);
    pl[c] = v; s += v; s2 += v * v;
  }
  red[tid] = s; __syncthreads();
  for (int st = 128; st > 0; st >>= 1) { if (tid < st) red[tid] += red[tid + st]; __syncthreads(); }
  float mu = red[0] * (1.f / CCH);
  __syncthreads();
  red[tid] = s2; __syncthreads();
  for (int st = 128; st > 0; st >>= 1) { if (tid < st) red[tid] += red[tid + st]; __syncthreads(); }
  float var = red[0] * (1.f / CCH) - mu * mu;
  float rstd = rsqrtf(var + 1e-5f);
  __syncthreads();
  for (int c = tid; c < CCH; c += 256) pl[c] = (pl[c] - mu) * rstd * lng[c] + lnb[c];
  __syncthreads();
  for (int o = tid; o < CCH; o += 256) {
    float a = b1[o];
    for (int c = 0; c < CCH; c++) a += w1[o * CCH + c] * pl[c];
    hb[o] = fmaxf(a, 0.f);
  }
  __syncthreads();
  float t = hb[tid] * w2[tid] + hb[tid + 256] * w2[tid + 256];
  red[tid] = t; __syncthreads();
  for (int st = 128; st > 0; st >>= 1) { if (tid < st) red[tid] += red[tid + st]; __syncthreads(); }
  if (tid == 0) out[b] = red[0] + b2[0];
}

// ---------------------------------------------------------------------------
extern "C" void kernel_launch(void* const* d_in, const int* in_sizes, int n_in,
                              void* d_out, int out_size, void* d_ws, size_t ws_size,
                              hipStream_t stream) {
  const float* x      = (const float*)d_in[0];
  const float* mlp_w1 = (const float*)d_in[1];
  const float* mlp_b1 = (const float*)d_in[2];
  const float* mlp_w2 = (const float*)d_in[3];
  const float* mlp_b2 = (const float*)d_in[4];
  const float* sp_w   = (const float*)d_in[5];
  const float* sp_g   = (const float*)d_in[6];
  const float* sp_b   = (const float*)d_in[7];
  const float* sp_m   = (const float*)d_in[8];
  const float* sp_v   = (const float*)d_in[9];
  const float* wq     = (const float*)d_in[10];
  const float* bq     = (const float*)d_in[11];
  const float* wk     = (const float*)d_in[12];
  const float* bk     = (const float*)d_in[13];
  const float* wv     = (const float*)d_in[14];
  const float* bv     = (const float*)d_in[15];
  const float* wo     = (const float*)d_in[16];
  const float* bo     = (const float*)d_in[17];
  const float* ln_g   = (const float*)d_in[18];
  const float* ln_b   = (const float*)d_in[19];
  const float* fc1_w  = (const float*)d_in[20];
  const float* fc1_b  = (const float*)d_in[21];
  const float* fc2_w  = (const float*)d_in[22];
  const float* fc2_b  = (const float*)d_in[23];
  float* out = (float*)d_out;

  // ---- workspace layout ----
  char* ws = (char*)d_ws;
  size_t off = 0;
  auto alloc = [&](size_t bytes) -> char* {
    char* p = ws + off;
    off += (bytes + 255) & ~(size_t)255;
    return p;
  };
  float* avg_p  = (float*)alloc((size_t)BSZ * CCH * 4);
  float* max_p  = (float*)alloc((size_t)BSZ * CCH * 4);
  float* ch     = (float*)alloc((size_t)BSZ * CCH * 4);
  float* cmax   = (float*)alloc((size_t)BSZ * NTOK * 4);
  float* cmean  = (float*)alloc((size_t)BSZ * NTOK * 4);
  float* sig    = (float*)alloc((size_t)BSZ * NTOK * 4);
  float* xg     = (float*)alloc((size_t)BSZ * CCH * NTOK * 4);
  bf16*  xbf    = (bf16*) alloc((size_t)BSZ * NTOK * CCH * 2);
  bf16*  Qbf    = (bf16*) alloc((size_t)BSZ * NTOK * CCH * 2);
  bf16*  Kbf    = (bf16*) alloc(((size_t)BSZ * NTOK + 32) * CCH * 2);  // +32 pad rows
  size_t vt_bytes = (size_t)BSZ * NHEAD * HDIM * NPAD * 2;
  bf16*  Vt     = (bf16*) alloc(vt_bytes);
  bf16*  Attn   = (bf16*) alloc((size_t)BSZ * NTOK * CCH * 2);
  float* pooled = (float*)alloc((size_t)BSZ * CCH * 4);
  bf16*  wqb    = (bf16*) alloc((size_t)CCH * CCH * 2);
  bf16*  wkb    = (bf16*) alloc((size_t)CCH * CCH * 2);
  bf16*  wvb    = (bf16*) alloc((size_t)CCH * CCH * 2);
  bf16*  wob    = (bf16*) alloc((size_t)CCH * CCH * 2);

  hipMemsetAsync(Vt, 0, vt_bytes, stream);                      // zero-pad keys 784..799
  hipMemsetAsync(pooled, 0, (size_t)BSZ * CCH * 4, stream);

  // weight conversion to bf16
  {
    int n = CCH * CCH, g = (n + 255) / 256;
    cvt_f32_bf16<<<g, 256, 0, stream>>>(wq, wqb, n);
    cvt_f32_bf16<<<g, 256, 0, stream>>>(wk, wkb, n);
    cvt_f32_bf16<<<g, 256, 0, stream>>>(wv, wvb, n);
    cvt_f32_bf16<<<g, 256, 0, stream>>>(wo, wob, n);
  }

  pool_avgmax<<<BSZ * CCH, 256, 0, stream>>>(x, avg_p, max_p);
  channel_mlp<<<BSZ, 256, 0, stream>>>(avg_p, max_p, mlp_w1, mlp_b1, mlp_w2, mlp_b2, ch);
  scale_comp<<<dim3(4, BSZ), 256, 0, stream>>>(x, ch, xg, cmax, cmean);
  spatial_sig<<<dim3(4, BSZ), 256, 0, stream>>>(cmax, cmean, sp_w, sp_g, sp_b, sp_m, sp_v, sig);
  gate_transpose<<<dim3(NTOK / 16, CCH / 16, BSZ), 256, 0, stream>>>(xg, sig, xbf);
  qkv_proj<<<dim3(NTOK / 16, 4, BSZ), 256, 0, stream>>>(xbf, wqb, wkb, wvb, bq, bk, bv,
                                                        Qbf, Kbf, Vt);
  flash_attn<<<dim3(7, NHEAD, BSZ), 256, 0, stream>>>(Qbf, Kbf, Vt, Attn);
  out_proj<<<dim3(NTOK / 16, 4, BSZ), 256, 0, stream>>>(Attn, wob, bo, xg, pooled);
  classifier<<<BSZ, 256, 0, stream>>>(pooled, ln_g, ln_b, fc1_w, fc1_b, fc2_w, fc2_b, out);
}
